// ContrastiveRetriever_12051678233106
// MI455X (gfx1250) — matrix operand order repaired and checked by reading, not compiled, not run on hardware
//
#include <hip/hip_runtime.h>
#include <hip/hip_bf16.h>
#include <stdint.h>

// ---------------------------------------------------------------------------
// Types for CDNA5 WMMA (wave32): v_wmma_f32_16x16x32_bf16
// ---------------------------------------------------------------------------
typedef __attribute__((ext_vector_type(16))) __bf16 v16bf;
typedef __attribute__((ext_vector_type(8)))  float  v8f;

__device__ __forceinline__ unsigned int f2bf_bits(float f) {
  unsigned int u = __float_as_uint(f);
  u += 0x7FFFu + ((u >> 16) & 1u);          // round-to-nearest-even
  return u >> 16;
}
__device__ __forceinline__ unsigned short f2bf(float f) {
  return (unsigned short)f2bf_bits(f);
}

// A-matrix fragment (16x32 bf16): lane<16 -> row M=lane, K={0..7,16..23};
// lane>=16 -> row M=lane-16, K={8..15,24..31}.  Two 16B chunks per lane.
__device__ __forceinline__ v16bf fragA(const unsigned short* row, int kbase) {
  union { uint4 q[2]; v16bf v; } u;
  u.q[0] = *(const uint4*)(row + kbase);
  u.q[1] = *(const uint4*)(row + kbase + 16);
  return u.v;
}
// B-matrix fragment (32x16 bf16), LDS stored [N][K]: lane<16 -> col N=lane,
// K={0..7,16..23}; lane>=16 -> col N=lane-16, K={8..15,24..31}.  Two 16B chunks per lane.
__device__ __forceinline__ v16bf fragB(const unsigned short* row, int kb) {
  union { uint4 q[2]; v16bf v; } u;
  u.q[0] = *(const uint4*)(row + kb);
  u.q[1] = *(const uint4*)(row + kb + 16);
  return u.v;
}

// ---------------------------------------------------------------------------
// Weight pre-pass: Wt[m*K + k] = bf16(W[k*M + m])   (transpose + convert once)
// ---------------------------------------------------------------------------
__global__ void convert_wT(const float* __restrict__ W, unsigned short* __restrict__ Wt,
                           int K, int M)
{
  int i = blockIdx.x * blockDim.x + threadIdx.x;
  if (i >= K * M) return;
  int m = i / K, k = i - m * K;
  Wt[i] = f2bf(W[(size_t)k * M + m]);
}

// ---------------------------------------------------------------------------
// Tiled WMMA GEMM:  Y[N,M] = act(X[N,K] @ W[K,M] + bias)
//   X: f32 row-major, Wt: bf16 [M][K] (pre-transposed), Y: f32.
// Block tile 128x128, BK=32, 256 threads = 8 waves (2x4 grid), each wave owns
// a 64x32 subtile (4x2 WMMA accumulators -> 8 wmma per k-step per wave).
// A tile staged with float4 loads + packed ds_store_b64; B tile staged with
// CDNA5 async global->LDS copies (weights are already bf16).
// Requires N%128==0, M%128==0, K%32==0 (true at every call site).
// ---------------------------------------------------------------------------
__global__ __launch_bounds__(256)
void gemm_bf16_wmma(const float* __restrict__ X, const unsigned short* __restrict__ Wt,
                    const float* __restrict__ bias, float* __restrict__ Y,
                    int N, int K, int M, int act)
{
  __shared__ __align__(16) unsigned short lA[128 * 32];   // [row][k]
  __shared__ __align__(16) unsigned short lB[128 * 32];   // [n][k]

  const int t    = threadIdx.x;
  const int lane = t & 31;
  const int wave = t >> 5;                  // 0..7
  const int row0 = blockIdx.x * 128;
  const int col0 = blockIdx.y * 128;
  const int wr   = (wave >> 2) * 64;        // wave row offset in tile (0/64)
  const int wc   = (wave & 3) * 32;         // wave col offset in tile (0..96)
  const int l16  = lane & 15;
  const int hi   = lane >> 4;               // 0 or 1
  const int kbA  = hi * 8;                  // A k-base per lane half
  const int kbB  = hi * 8;                  // B k-base per lane half

  v8f acc[4][2];
#pragma unroll
  for (int i = 0; i < 4; ++i)
#pragma unroll
    for (int j = 0; j < 2; ++j)
#pragma unroll
      for (int r = 0; r < 8; ++r) acc[i][j][r] = 0.0f;

  for (int k0 = 0; k0 < K; k0 += 32) {
    __syncthreads();
    // ---- stage A tile 128x32: float4 load, pack to 4xbf16, ds_store_b64 ----
#pragma unroll
    for (int i = t * 4; i < 128 * 32; i += 256 * 4) {
      int r = i >> 5, c = i & 31;
      float4 v = *(const float4*)(X + (size_t)(row0 + r) * K + (k0 + c));
      uint2 p;
      p.x = f2bf_bits(v.x) | (f2bf_bits(v.y) << 16);
      p.y = f2bf_bits(v.z) | (f2bf_bits(v.w) << 16);
      *(uint2*)(lA + i) = p;
    }
    // ---- stage B tile 128x32: async global->LDS byte copy (already bf16) ----
#pragma unroll
    for (int i = t * 4; i < 128 * 32; i += 256 * 4) {
      int n = i >> 5, c = i & 31;
      unsigned long long g =
          (unsigned long long)(uintptr_t)(Wt + (size_t)(col0 + n) * K + (k0 + c));
      unsigned int lds = (unsigned int)(uintptr_t)(lB + i);
      asm volatile("global_load_async_to_lds_b64 %0, %1, off"
                   :: "v"(lds), "v"(g) : "memory");
    }
    asm volatile("s_wait_asynccnt 0x0" ::: "memory");
    __syncthreads();

    // ---- fragments + 8 WMMAs per wave ----
    v16bf aF[4], bF[2];
#pragma unroll
    for (int mi = 0; mi < 4; ++mi)
      aF[mi] = fragA(&lA[(wr + mi * 16 + l16) * 32], kbA);
#pragma unroll
    for (int ni = 0; ni < 2; ++ni)
      bF[ni] = fragB(&lB[(wc + ni * 16 + l16) * 32], kbB);

#pragma unroll
    for (int mi = 0; mi < 4; ++mi)
#pragma unroll
      for (int ni = 0; ni < 2; ++ni)
        acc[mi][ni] = __builtin_amdgcn_wmma_f32_16x16x32_bf16(
            false, aF[mi], false, bF[ni], (short)0, acc[mi][ni], false, false);
  }

  // C/D layout: vgpr r -> M = r + 8*hi, N = lane%16
#pragma unroll
  for (int mi = 0; mi < 4; ++mi)
#pragma unroll
    for (int ni = 0; ni < 2; ++ni) {
      const int col = col0 + wc + ni * 16 + l16;
      const float bv = bias ? bias[col] : 0.0f;
#pragma unroll
      for (int r = 0; r < 8; ++r) {
        const int row = row0 + wr + mi * 16 + r + hi * 8;
        float v = acc[mi][ni][r] + bv;
        if (act) v = fmaxf(v, 0.0f);
        Y[(size_t)row * M + col] = v;
      }
    }
}

// ---------------------------------------------------------------------------
// Elementwise / graph kernels (fp32, hidden dim fixed to 128 for GCN layers)
// ---------------------------------------------------------------------------
__global__ void fill_zero(float* __restrict__ p, int n) {
  int i = blockIdx.x * blockDim.x + threadIdx.x;
  if (i < n) p[i] = 0.0f;
}

__global__ void deg_edges(const int* __restrict__ dst, int E, float* __restrict__ deg) {
  int e = blockIdx.x * blockDim.x + threadIdx.x;
  if (e < E) atomicAdd(&deg[dst[e]], 1.0f);
}

__global__ void deg_rsqrt(float* __restrict__ deg, int N) {
  int n = blockIdx.x * blockDim.x + threadIdx.x;
  if (n < N) deg[n] = rsqrtf(deg[n] + 1.0f);   // +1 for the self-loop
}

// out[dst] += H[src] * dis[src]*dis[dst]; one lane per (edge, 4 channels)
__global__ void scatter_edges(const float* __restrict__ H, const int* __restrict__ src,
                              const int* __restrict__ dst, const float* __restrict__ dis,
                              float* __restrict__ out, int E)
{
  int g = blockIdx.x * blockDim.x + threadIdx.x;
  int e = g >> 5;
  if (e >= E) return;
  int c = (g & 31) * 4;
  int s = src[e], d = dst[e];
  float w = dis[s] * dis[d];
  float4 h = *(const float4*)(H + (size_t)s * 128 + c);
  float* o = out + (size_t)d * 128 + c;
  atomicAdd(o + 0, h.x * w);
  atomicAdd(o + 1, h.y * w);
  atomicAdd(o + 2, h.z * w);
  atomicAdd(o + 3, h.w * w);
}

// In-place: H[n] = relu(SC[n] + H[n]*dis[n]^2 + bias)   (self-loop folded in)
__global__ void combine_self_bias_relu(float* __restrict__ H, const float* __restrict__ SC,
                                       const float* __restrict__ dis,
                                       const float* __restrict__ bias, int N)
{
  int g = blockIdx.x * blockDim.x + threadIdx.x;
  int n = g >> 5;
  if (n >= N) return;
  int c = (g & 31) * 4;
  float d2 = dis[n] * dis[n];
  float4 h  = *(const float4*)(H  + (size_t)n * 128 + c);
  float4 sc = *(const float4*)(SC + (size_t)n * 128 + c);
  float4 r;
  r.x = fmaxf(sc.x + h.x * d2 + bias[c + 0], 0.0f);
  r.y = fmaxf(sc.y + h.y * d2 + bias[c + 1], 0.0f);
  r.z = fmaxf(sc.z + h.z * d2 + bias[c + 2], 0.0f);
  r.w = fmaxf(sc.w + h.w * d2 + bias[c + 3], 0.0f);
  *(float4*)(H + (size_t)n * 128 + c) = r;
}

__global__ void pool_sum(const float* __restrict__ H, const int* __restrict__ batch,
                         float* __restrict__ sums, int N)
{
  int g = blockIdx.x * blockDim.x + threadIdx.x;
  int n = g >> 5;
  if (n >= N) return;
  int c = (g & 31) * 4;
  int b = batch[n];
  float4 h = *(const float4*)(H + (size_t)n * 128 + c);
  float* o = sums + (size_t)b * 128 + c;
  atomicAdd(o + 0, h.x);
  atomicAdd(o + 1, h.y);
  atomicAdd(o + 2, h.z);
  atomicAdd(o + 3, h.w);
}

__global__ void pool_cnt(const int* __restrict__ batch, float* __restrict__ cnt, int N) {
  int n = blockIdx.x * blockDim.x + threadIdx.x;
  if (n < N) atomicAdd(&cnt[batch[n]], 1.0f);
}

__global__ void pool_div(const float* __restrict__ sums, const float* __restrict__ cnt,
                         float* __restrict__ pooled, int G)
{
  int g = blockIdx.x * blockDim.x + threadIdx.x;  // G*128 threads
  int gi = g >> 7;
  if (gi >= G) return;
  pooled[g] = sums[g] / fmaxf(cnt[gi], 1.0f);
}

// ---------------------------------------------------------------------------
// Host orchestration
// ---------------------------------------------------------------------------
namespace {
constexpr int N_NODES  = 524288;
constexpr int N_GRAPHS = 16384;
constexpr int N_EDGES  = 2097152;
constexpr int FEAT = 64, HID = 128, OUT = 256, DREAMS = 1024, PROJ_HID = 512;

inline void gemm_launch(const float* X, const unsigned short* Wt, const float* b,
                        float* Y, int N, int K, int M, int act, hipStream_t s) {
  gemm_bf16_wmma<<<dim3(N / 128, M / 128), 256, 0, s>>>(X, Wt, b, Y, N, K, M, act);
}

void run_encoder(const float* x, const int* edge, const int* batch,
                 const unsigned short* w0, const float* b0,
                 const unsigned short* w1, const float* b1,
                 const unsigned short* w2, const float* b2,
                 const unsigned short* fcw, const float* fcb,
                 float* out, float* bufA, float* bufB, float* dis,
                 float* sums, float* cnt, float* pooled, hipStream_t s)
{
  const int N = N_NODES, E = N_EDGES, G = N_GRAPHS;
  const int* src = edge;
  const int* dst = edge + E;
  const int NH = N * HID;

  // symmetric norm: dis = 1/sqrt(deg_in + 1)
  fill_zero<<<(N + 255) / 256, 256, 0, s>>>(dis, N);
  deg_edges<<<(E + 255) / 256, 256, 0, s>>>(dst, E, dis);
  deg_rsqrt<<<(N + 255) / 256, 256, 0, s>>>(dis, N);

  // layer 0: H=bufA, scatter->bufB, x1 in bufA
  gemm_launch(x, w0, nullptr, bufA, N, FEAT, HID, 0, s);
  fill_zero<<<(NH + 255) / 256, 256, 0, s>>>(bufB, NH);
  scatter_edges<<<(E * 32 + 255) / 256, 256, 0, s>>>(bufA, src, dst, dis, bufB, E);
  combine_self_bias_relu<<<(N * 32 + 255) / 256, 256, 0, s>>>(bufA, bufB, dis, b0, N);

  // layer 1: H=bufB, scatter->bufA, x2 in bufB
  gemm_launch(bufA, w1, nullptr, bufB, N, HID, HID, 0, s);
  fill_zero<<<(NH + 255) / 256, 256, 0, s>>>(bufA, NH);
  scatter_edges<<<(E * 32 + 255) / 256, 256, 0, s>>>(bufB, src, dst, dis, bufA, E);
  combine_self_bias_relu<<<(N * 32 + 255) / 256, 256, 0, s>>>(bufB, bufA, dis, b1, N);

  // layer 2: H=bufA, scatter->bufB, x3 in bufA
  gemm_launch(bufB, w2, nullptr, bufA, N, HID, HID, 0, s);
  fill_zero<<<(NH + 255) / 256, 256, 0, s>>>(bufB, NH);
  scatter_edges<<<(E * 32 + 255) / 256, 256, 0, s>>>(bufA, src, dst, dis, bufB, E);
  combine_self_bias_relu<<<(N * 32 + 255) / 256, 256, 0, s>>>(bufA, bufB, dis, b2, N);

  // global mean pool + fc
  fill_zero<<<(G * HID + 255) / 256, 256, 0, s>>>(sums, G * HID);
  fill_zero<<<(G + 255) / 256, 256, 0, s>>>(cnt, G);
  pool_sum<<<(N * 32 + 255) / 256, 256, 0, s>>>(bufA, batch, sums, N);
  pool_cnt<<<(N + 255) / 256, 256, 0, s>>>(batch, cnt, N);
  pool_div<<<(G * HID + 255) / 256, 256, 0, s>>>(sums, cnt, pooled, G);
  gemm_launch(pooled, fcw, fcb, out, G, HID, OUT, 0, s);
}
} // namespace

extern "C" void kernel_launch(void* const* d_in, const int* in_sizes, int n_in,
                              void* d_out, int out_size, void* d_ws, size_t ws_size,
                              hipStream_t stream)
{
  const float* dreams   = (const float*)d_in[0];
  const float* pos_x    = (const float*)d_in[1];
  const int*   pos_ei   = (const int*)d_in[2];
  const int*   pos_b    = (const int*)d_in[3];
  const float* neg_x    = (const float*)d_in[4];
  const int*   neg_ei   = (const int*)d_in[5];
  const int*   neg_b    = (const int*)d_in[6];
  const float* p_w1 = (const float*)d_in[7];   const float* p_b1 = (const float*)d_in[8];
  const float* p_w2 = (const float*)d_in[9];   const float* p_b2 = (const float*)d_in[10];
  const float* g_w0 = (const float*)d_in[11];  const float* g_b0 = (const float*)d_in[12];
  const float* g_w1 = (const float*)d_in[13];  const float* g_b1 = (const float*)d_in[14];
  const float* g_w2 = (const float*)d_in[15];  const float* g_b2 = (const float*)d_in[16];
  const float* fc_w = (const float*)d_in[17];  const float* fc_b = (const float*)d_in[18];

  float* out_anchor = (float*)d_out;                           // [16384,256]
  float* out_pos    = out_anchor + (size_t)N_GRAPHS * OUT;
  float* out_neg    = out_pos    + (size_t)N_GRAPHS * OUT;

  // workspace layout (floats)
  float* bufA   = (float*)d_ws;                     // 524288*128
  float* bufB   = bufA + (size_t)N_NODES * HID;     // 524288*128
  float* dis    = bufB + (size_t)N_NODES * HID;     // 524288
  float* sums   = dis  + N_NODES;                   // 16384*128
  float* cnt    = sums + (size_t)N_GRAPHS * HID;    // 16384
  float* pooled = cnt  + N_GRAPHS;                  // 16384*128
  // bf16 transposed weights (ushort)
  unsigned short* wt_p1 = (unsigned short*)(pooled + (size_t)N_GRAPHS * HID);
  unsigned short* wt_p2 = wt_p1 + (size_t)DREAMS * PROJ_HID;   // 1024*512
  unsigned short* wt_g0 = wt_p2 + (size_t)PROJ_HID * OUT;      // 512*256
  unsigned short* wt_g1 = wt_g0 + (size_t)FEAT * HID;          // 64*128
  unsigned short* wt_g2 = wt_g1 + (size_t)HID * HID;           // 128*128
  unsigned short* wt_fc = wt_g2 + (size_t)HID * HID;           // 128*128 -> 128*256

  // ---- one-time (per call) weight transpose + bf16 convert ----
  auto cvt = [&](const float* W, unsigned short* Wt, int K, int M) {
    convert_wT<<<(K * M + 255) / 256, 256, 0, stream>>>(W, Wt, K, M);
  };
  cvt(p_w1, wt_p1, DREAMS, PROJ_HID);
  cvt(p_w2, wt_p2, PROJ_HID, OUT);
  cvt(g_w0, wt_g0, FEAT, HID);
  cvt(g_w1, wt_g1, HID, HID);
  cvt(g_w2, wt_g2, HID, HID);
  cvt(fc_w, wt_fc, HID, OUT);

  // ---- anchor: 2-layer MLP (uses bufA as hidden) ----
  gemm_launch(dreams, wt_p1, p_b1, bufA, N_GRAPHS, DREAMS, PROJ_HID, 1, stream);
  gemm_launch(bufA, wt_p2, p_b2, out_anchor, N_GRAPHS, PROJ_HID, OUT, 0, stream);

  // ---- positive / negative GNN encoders ----
  run_encoder(pos_x, pos_ei, pos_b, wt_g0, g_b0, wt_g1, g_b1, wt_g2, g_b2,
              wt_fc, fc_b, out_pos, bufA, bufB, dis, sums, cnt, pooled, stream);
  run_encoder(neg_x, neg_ei, neg_b, wt_g0, g_b0, wt_g1, g_b1, wt_g2, g_b2,
              wt_fc, fc_b, out_neg, bufA, bufB, dis, sums, cnt, pooled, stream);

  (void)in_sizes; (void)n_in; (void)out_size; (void)ws_size;
}
